// GCN_40132174414142
// MI455X (gfx1250) — matrix operand order repaired and checked by reading, not compiled, not run on hardware
//
#include <hip/hip_runtime.h>
#include <hip/hip_bf16.h>

typedef __bf16 bf16_t;
typedef __attribute__((ext_vector_type(16))) __bf16 v16bf;
typedef __attribute__((ext_vector_type(8)))  float  v8f;

#define NEG_SLOPE 0.2f

__device__ __forceinline__ void atomAddF(float* p, float v) {
    __hip_atomic_fetch_add(p, v, __ATOMIC_RELAXED, __HIP_MEMORY_SCOPE_AGENT);
}

// ---------------------------------------------------------------------------
// Utility kernels
// ---------------------------------------------------------------------------
__global__ void zero_kernel(float* __restrict__ p, size_t n) {
    size_t i = (size_t)blockIdx.x * blockDim.x + threadIdx.x;
    size_t stride = (size_t)gridDim.x * blockDim.x;
    for (; i < n; i += stride) p[i] = 0.0f;
}

__global__ void degree_kernel(const int* __restrict__ src, const int* __restrict__ dst,
                              float* __restrict__ odeg, float* __restrict__ ideg, int E) {
    int e = blockIdx.x * blockDim.x + threadIdx.x;
    if (e < E) {
        atomAddF(&odeg[src[e]], 1.0f);
        atomAddF(&ideg[dst[e]], 1.0f);
    }
}

__global__ void norm_finalize_kernel(float* __restrict__ p, size_t n) {
    size_t i = (size_t)blockIdx.x * blockDim.x + threadIdx.x;
    if (i < n) {
        float v = p[i];
        v = v < 1.0f ? 1.0f : v;
        p[i] = rsqrtf(v);
    }
}

// One-time weight preprocessing: W[K][N] f32 row-major -> Wt[Np][Kp] bf16,
// transposed (Wt[n][k] = W[k][n]) and zero-padded to Np=rup(N,64), Kp=rup(K,32).
__global__ void wt_convert_kernel(const float* __restrict__ W, bf16_t* __restrict__ Wt,
                                  int K, int N, int Kp, int Np) {
    int i = blockIdx.x * blockDim.x + threadIdx.x;
    int total = Np * Kp;
    if (i >= total) return;
    int n = i / Kp;
    int k = i % Kp;
    float v = (n < N && k < K) ? W[(size_t)k * N + n] : 0.0f;
    Wt[i] = (bf16_t)v;
}

// wave-per-edge scatter-add: out[dst[e]] += feat[src[e]] * w[e] * (onorm?onorm[src[e]]:1)
// F must be a multiple of 4.
__global__ void agg_kernel(const float* __restrict__ feat, float* __restrict__ out,
                           const int* __restrict__ src, const int* __restrict__ dst,
                           const float* __restrict__ w, const float* __restrict__ onorm,
                           int E, int F) {
    const int wv = threadIdx.x >> 5;
    const int lane = threadIdx.x & 31;
    const int e = blockIdx.x * (blockDim.x >> 5) + wv;
    if (e >= E) return;
    const int s = src[e];
    const int d = dst[e];
    float sc = w[e];
    if (onorm) sc *= onorm[s];
    const float* fs = feat + (size_t)s * F;
    float* od = out + (size_t)d * F;
    for (int f = lane * 4; f < F; f += 128) {
        float4 v = *(const float4*)(fs + f);
        atomAddF(&od[f + 0], v.x * sc);
        atomAddF(&od[f + 1], v.y * sc);
        atomAddF(&od[f + 2], v.z * sc);
        atomAddF(&od[f + 3], v.w * sc);
    }
}

// out = in * inorm[row] + bias[col]  (optional relu)
__global__ void postagg_kernel(const float* __restrict__ in, float* __restrict__ out,
                               const float* __restrict__ inorm, const float* __restrict__ bias,
                               int rows, int F, int doRelu) {
    size_t i = (size_t)blockIdx.x * blockDim.x + threadIdx.x;
    size_t total = (size_t)rows * F;
    if (i >= total) return;
    int row = (int)(i / F);
    int col = (int)(i % F);
    float v = in[i] * inorm[row] + bias[col];
    if (doRelu) v = v > 0.0f ? v : 0.0f;
    out[i] = v;
}

// out[i] = dot(z[i,:K], P3) + pb3   (K multiple of 4; rows 16B-aligned)
__global__ void dot_final_kernel(const float* __restrict__ z, const float* __restrict__ P3,
                                 const float* __restrict__ pb3, float* __restrict__ out,
                                 int M, int K) {
    int i = blockIdx.x * blockDim.x + threadIdx.x;
    if (i >= M) return;
    const float* zi = z + (size_t)i * K;
    float acc = pb3[0];
#pragma unroll 4
    for (int k = 0; k < K; k += 4) {
        float4 v = *(const float4*)(zi + k);
        float4 p = *(const float4*)(P3 + k);
        acc += v.x * p.x + v.y * p.y + v.z * p.z + v.w * p.w;
    }
    out[i] = acc;
}

// ---------------------------------------------------------------------------
// bf16 WMMA GEMM (LDS-free, barrier-free, K fully unrolled at compile time):
//   D[M,N] = epilogue( A'[M,K] @ B[K,N] )
//   A' = A * rowScale[row]            (GATHER==false)
//   A' = A[gs[row],:] * A[gd[row],:]  (GATHER==true, elementwise product)
//   B supplied as Wt[Np][KP] bf16, pre-transposed + zero padded: each fragment
//   is two aligned 16B global loads already in the ISA WMMA B layout
//   (lane: col=nt*16+lm, K=16*(j/8)+half*8+j%8). Weights are tiny and L2/WGP$-resident.
// Block: 256 threads = 8 waves; block tile 128x64; each wave: 16 rows x 64
// cols = 4 WMMA/K-step sharing one A fragment (ISA 16-bit A 16x32 layout).
// K, KP are template constants: the K-loop fully unrolls, tail guards fold
// statically (all K are multiples of 16 -> each 16-wide A chunk is either
// fully valid or fully zero), and no cross-iteration register copies remain.
// ---------------------------------------------------------------------------
template <int EPI, bool GATHER, int K, int KP>
__global__ __launch_bounds__(256) void gemm_bf16_wmma(
    const float* __restrict__ A, int lda,
    const bf16_t* __restrict__ Wt,
    float* __restrict__ D, int ldd,
    int M, int N,
    const float* __restrict__ rowScale,
    const float* __restrict__ epiRow,
    const float* __restrict__ bias,
    const int* __restrict__ gs, const int* __restrict__ gd) {
    const int tid = threadIdx.x;
    const int rowBase = blockIdx.y * 128;
    const int colBase = blockIdx.x * 64;
    const int wv = tid >> 5;
    const int lane = tid & 31;
    const int lm = lane & 15;
    const int half = lane >> 4;

    v8f c[4] = {{}, {}, {}, {}};

    // A row for this lane (clamped; out-of-range rows never stored)
    const int aRow = rowBase + wv * 16 + lm;
    const int rowC = aRow < M ? aRow : (M - 1);
    const float* Arow;
    const float* Arow2 = nullptr;
    float rs = 1.0f;
    if (GATHER) {
        Arow = A + (size_t)gs[rowC] * lda;
        Arow2 = A + (size_t)gd[rowC] * lda;
    } else {
        Arow = A + (size_t)rowC * lda;
        if (rowScale) rs = rowScale[rowC];
    }

    // Per-lane B fragment base pointers (16B-aligned: KP%32==0)
    const bf16_t* Bp[4];
#pragma unroll
    for (int nt = 0; nt < 4; ++nt)
        Bp[nt] = Wt + (size_t)(colBase + nt * 16 + lm) * KP + half * 8;

#pragma unroll
    for (int k0 = 0; k0 < KP; k0 += 32) {
        // ---- A fragment direct from global (chunks statically valid or zero)
        v16bf a;
        const int kb = k0 + half * 8;
        if (k0 + 16 <= K) {  // chunk 1: K range [k0, k0+16)
            float4 p0 = *(const float4*)(Arow + kb);
            float4 p1 = *(const float4*)(Arow + kb + 4);
            if (GATHER) {
                float4 q0 = *(const float4*)(Arow2 + kb);
                float4 q1 = *(const float4*)(Arow2 + kb + 4);
                p0.x *= q0.x; p0.y *= q0.y; p0.z *= q0.z; p0.w *= q0.w;
                p1.x *= q1.x; p1.y *= q1.y; p1.z *= q1.z; p1.w *= q1.w;
            }
            a[0] = (bf16_t)(p0.x * rs); a[1] = (bf16_t)(p0.y * rs);
            a[2] = (bf16_t)(p0.z * rs); a[3] = (bf16_t)(p0.w * rs);
            a[4] = (bf16_t)(p1.x * rs); a[5] = (bf16_t)(p1.y * rs);
            a[6] = (bf16_t)(p1.z * rs); a[7] = (bf16_t)(p1.w * rs);
        } else {
#pragma unroll
            for (int j = 0; j < 8; ++j) a[j] = (bf16_t)0.0f;
        }
        if (k0 + 32 <= K) {  // chunk 2: K range [k0+16, k0+32)
            float4 p2 = *(const float4*)(Arow + kb + 16);
            float4 p3 = *(const float4*)(Arow + kb + 20);
            if (GATHER) {
                float4 q2 = *(const float4*)(Arow2 + kb + 16);
                float4 q3 = *(const float4*)(Arow2 + kb + 20);
                p2.x *= q2.x; p2.y *= q2.y; p2.z *= q2.z; p2.w *= q2.w;
                p3.x *= q3.x; p3.y *= q3.y; p3.z *= q3.z; p3.w *= q3.w;
            }
            a[8]  = (bf16_t)(p2.x * rs); a[9]  = (bf16_t)(p2.y * rs);
            a[10] = (bf16_t)(p2.z * rs); a[11] = (bf16_t)(p2.w * rs);
            a[12] = (bf16_t)(p3.x * rs); a[13] = (bf16_t)(p3.y * rs);
            a[14] = (bf16_t)(p3.z * rs); a[15] = (bf16_t)(p3.w * rs);
        } else {
#pragma unroll
            for (int j = 8; j < 16; ++j) a[j] = (bf16_t)0.0f;
        }

        // ---- 4 WMMAs reusing the A fragment
#pragma unroll
        for (int nt = 0; nt < 4; ++nt) {
            typedef __attribute__((ext_vector_type(8))) __bf16 v8bf_t;
            v8bf_t blo = *(const v8bf_t*)(Bp[nt] + k0);
            v8bf_t bhi = *(const v8bf_t*)(Bp[nt] + k0 + 16);
            v16bf b = __builtin_shufflevector(blo, bhi, 0, 1, 2, 3, 4, 5, 6, 7, 8, 9, 10, 11, 12, 13, 14, 15);
            c[nt] = __builtin_amdgcn_wmma_f32_16x16x32_bf16(false, a, false, b,
                                                            (short)0, c[nt], false, false);
        }
    }

    // ---- epilogue: C/D layout = VGPR r, lanes 0-15 -> M=r, lanes 16-31 -> M=8+r
    const int colL = colBase + lm;
#pragma unroll
    for (int r = 0; r < 8; ++r) {
        const int orow = rowBase + wv * 16 + half * 8 + r;
        if (orow >= M) continue;
#pragma unroll
        for (int nt = 0; nt < 4; ++nt) {
            const int col = colL + nt * 16;
            if (col >= N) continue;
            float v = c[nt][r];
            if (EPI == 1) {
                float t = v * epiRow[orow] + bias[col];
                D[(size_t)orow * ldd + col] = t > 0.0f ? t : 0.0f;
            } else if (EPI == 2) {
                float t = v + bias[col];
                D[(size_t)orow * ldd + col] = t > 0.0f ? t : NEG_SLOPE * t;
            } else {
                D[(size_t)orow * ldd + col] = v;
            }
        }
    }
}

// ---------------------------------------------------------------------------
// Host orchestration
// ---------------------------------------------------------------------------
static inline int rup(int x, int m) { return (x + m - 1) / m * m; }

extern "C" void kernel_launch(void* const* d_in, const int* in_sizes, int n_in,
                              void* d_out, int out_size, void* d_ws, size_t ws_size,
                              hipStream_t stream) {
    const float* x = (const float*)d_in[0];
    const float* w = (const float*)d_in[1];
    const int* src = (const int*)d_in[2];
    const int* dst = (const int*)d_in[3];
    const int* pos_src = (const int*)d_in[4];
    const int* pos_dst = (const int*)d_in[5];
    const int* neg_src = (const int*)d_in[6];
    const int* neg_dst = (const int*)d_in[7];
    const float* W1 = (const float*)d_in[8];
    const float* b1 = (const float*)d_in[9];
    const float* W2 = (const float*)d_in[10];
    const float* b2 = (const float*)d_in[11];
    const float* W3 = (const float*)d_in[12];
    const float* b3 = (const float*)d_in[13];
    const float* P1 = (const float*)d_in[14];
    const float* pb1 = (const float*)d_in[15];
    const float* P2 = (const float*)d_in[16];
    const float* pb2 = (const float*)d_in[17];
    const float* P3 = (const float*)d_in[18];
    const float* pb3 = (const float*)d_in[19];

    const int IN = 256, H1 = 512, H2 = 256, H3 = 160, PH1 = 80, PH2 = 40;
    const int N = in_sizes[0] / IN;   // 50000
    const int E = in_sizes[1];        // 800000
    const int EP = in_sizes[4];       // 200000

    // workspace layout
    float* ws = (float*)d_ws;
    float* onorm = ws;
    float* inorm = onorm + N;
    float* bufA = inorm + N;                 // N x 512
    float* bufB = bufA + (size_t)N * H1;     // N x 512
    float* z1 = bufA;                        // reuse after layer 3: EP x 80
    float* z2 = bufB;                        // EP x 40

    // padded transposed bf16 weights after the float buffers
    bf16_t* wt = (bf16_t*)(bufB + (size_t)N * H1);
    struct WtDesc { const float* W; int K, Nc, Kp, Np; bf16_t* p; };
    WtDesc wd[5] = {
        {W1, IN,  H1,  rup(IN, 32),  rup(H1, 64),  nullptr},   // 256x512
        {W2, H1,  H2,  rup(H1, 32),  rup(H2, 64),  nullptr},   // 512x256
        {W3, H2,  H3,  rup(H2, 32),  rup(H3, 64),  nullptr},   // 256x160->192
        {P1, H3,  PH1, rup(H3, 32),  rup(PH1, 64), nullptr},   // 160x80->128
        {P2, PH1, PH2, rup(PH1, 32), rup(PH2, 64), nullptr},   // 80->96 x 40->64
    };
    {
        bf16_t* p = wt;
        for (int i = 0; i < 5; ++i) {
            wd[i].p = p;
            p += (size_t)wd[i].Np * wd[i].Kp;
        }
    }

    float* out_pos = (float*)d_out;
    float* out_neg = out_pos + EP;
    float* h3 = out_neg + EP;  // N x 160, part of output

    auto zero = [&](float* p, size_t n) {
        int blocks = (int)((n + 256 * 8 - 1) / (256 * 8));
        if (blocks > 65535) blocks = 65535;
        zero_kernel<<<blocks, 256, 0, stream>>>(p, n);
    };

    // ---- one-time weight transpose/convert/pad (tiny)
    for (int i = 0; i < 5; ++i) {
        int total = wd[i].Np * wd[i].Kp;
        wt_convert_kernel<<<(total + 255) / 256, 256, 0, stream>>>(
            wd[i].W, wd[i].p, wd[i].K, wd[i].Nc, wd[i].Kp, wd[i].Np);
    }

    // ---- degree norms
    zero(onorm, (size_t)2 * N);
    degree_kernel<<<(E + 255) / 256, 256, 0, stream>>>(src, dst, onorm, inorm, E);
    norm_finalize_kernel<<<(2 * N + 255) / 256, 256, 0, stream>>>(onorm, (size_t)2 * N);

    dim3 blk(256);
    const int EDGES_PER_BLK = 8;  // 8 waves/block, wave-per-edge

    // ---- layer 1: agg = segsum((x*onorm)[src]*w) ; h1 = relu(agg@W1 * inorm + b1)
    zero(bufA, (size_t)N * IN);
    agg_kernel<<<(E + EDGES_PER_BLK - 1) / EDGES_PER_BLK, 256, 0, stream>>>(
        x, bufA, src, dst, w, onorm, E, IN);
    {
        dim3 grid((H1 + 63) / 64, (N + 127) / 128);
        gemm_bf16_wmma<1, false, 256, 256><<<grid, blk, 0, stream>>>(
            bufA, IN, wd[0].p, bufB, H1, N, H1,
            nullptr, inorm, b1, nullptr, nullptr);
    }

    // ---- layer 2: t2 = (h1*onorm)@W2 ; agg ; h2 = relu(agg*inorm + b2)
    {
        dim3 grid((H2 + 63) / 64, (N + 127) / 128);
        gemm_bf16_wmma<0, false, 512, 512><<<grid, blk, 0, stream>>>(
            bufB, H1, wd[1].p, bufA, H2, N, H2,
            onorm, nullptr, nullptr, nullptr, nullptr);
    }
    zero(bufB, (size_t)N * H2);
    agg_kernel<<<(E + EDGES_PER_BLK - 1) / EDGES_PER_BLK, 256, 0, stream>>>(
        bufA, bufB, src, dst, w, nullptr, E, H2);
    postagg_kernel<<<(int)(((size_t)N * H2 + 255) / 256), 256, 0, stream>>>(
        bufB, bufA, inorm, b2, N, H2, 1);

    // ---- layer 3: t3 = (h2*onorm)@W3 ; agg -> h3 ; h3 = h3*inorm + b3 (no relu)
    {
        dim3 grid((H3 + 63) / 64, (N + 127) / 128);
        gemm_bf16_wmma<0, false, 256, 256><<<grid, blk, 0, stream>>>(
            bufA, H2, wd[2].p, bufB, H3, N, H3,
            onorm, nullptr, nullptr, nullptr, nullptr);
    }
    zero(h3, (size_t)N * H3);
    agg_kernel<<<(E + EDGES_PER_BLK - 1) / EDGES_PER_BLK, 256, 0, stream>>>(
        bufB, h3, src, dst, w, nullptr, E, H3);
    postagg_kernel<<<(int)(((size_t)N * H3 + 255) / 256), 256, 0, stream>>>(
        h3, h3, inorm, b3, N, H3, 0);

    // ---- predictor on pos / neg pairs
    const int* pair_s[2] = {pos_src, neg_src};
    const int* pair_d[2] = {pos_dst, neg_dst};
    float* pair_out[2] = {out_pos, out_neg};
    for (int p = 0; p < 2; ++p) {
        {  // z1 = leaky(h3[s]*h3[d] @ P1 + pb1)   (gather-product fused in A load)
            dim3 grid((PH1 + 63) / 64, (EP + 127) / 128);
            gemm_bf16_wmma<2, true, 160, 160><<<grid, blk, 0, stream>>>(
                h3, H3, wd[3].p, z1, PH1, EP, PH1,
                nullptr, nullptr, pb1, pair_s[p], pair_d[p]);
        }
        {  // z2 = leaky(z1 @ P2 + pb2)  (K=80 -> KP=96: upper half of last step statically zeroed)
            dim3 grid((PH2 + 63) / 64, (EP + 127) / 128);
            gemm_bf16_wmma<2, false, 80, 96><<<grid, blk, 0, stream>>>(
                z1, PH1, wd[4].p, z2, PH2, EP, PH2,
                nullptr, nullptr, pb2, nullptr, nullptr);
        }
        dot_final_kernel<<<(EP + 255) / 256, 256, 0, stream>>>(
            z2, P3, pb3, pair_out[p], EP, PH2);
    }
    (void)n_in; (void)out_size; (void)ws_size;
}
